// IMMPNNWebshellClassifier_61469571940402
// MI455X (gfx1250) — matrix operand order repaired and checked by reading, not compiled, not run on hardware
//
#include <hip/hip_runtime.h>
#include <utility>

// Problem constants (match reference)
constexpr int Nn  = 100000;
constexpr int Ee  = 600000;
constexpr int Bb  = 50;
constexpr int Bpad = 64;       // Bb padded to multiple of 16 for unguarded A loads
constexpr int Ff  = 40;
constexpr int TFf = Bb * Ff;   // 2000
constexpr int D0c = 64;
constexpr int Hc  = 128;
constexpr int Cc  = 2;

typedef __attribute__((ext_vector_type(16))) __bf16 v16bf;
typedef __attribute__((ext_vector_type(8)))  float  v8f;

__device__ __forceinline__ unsigned short f2bfu(float f) {
    union { float f; unsigned u; } x; x.f = f;
    unsigned r = x.u + 0x7FFFu + ((x.u >> 16) & 1u);   // round-to-nearest-even
    return (unsigned short)(r >> 16);
}

__device__ __forceinline__ float bfu2f(unsigned short s) {
    union { unsigned u; float f; } y; y.u = ((unsigned)s) << 16;
    return y.f;
}

// One wave computes a full 16-row x 128-col output stripe (8 WMMA tiles).
// A fragment (ISA 7.12.2 16-bit A 16x32) is loaded once per k-step and reused
// across all 8 N-tiles; the 8 WMMAs per k-step are independent accumulators.
// Requires A to be readable for ceil16(M) rows (buffers are padded).
template <int KSTEPS, bool HASA2>
__global__ __launch_bounds__(256) void gemm_stripe_k(
    const unsigned short* __restrict__ A,  const unsigned short* __restrict__ Wa,
    const unsigned short* __restrict__ A2, const unsigned short* __restrict__ Wb,
    const unsigned short* __restrict__ addRows, const int* __restrict__ addIdx,
    const float* __restrict__ bias, unsigned short* __restrict__ out, int M, int doRelu)
{
    constexpr int K0 = KSTEPS * 32;
    const int lane   = threadIdx.x & 31;
    const int wave   = threadIdx.x >> 5;
    const int Mt     = (M + 15) >> 4;
    const int stripe = blockIdx.x * 8 + wave;
    if (stripe >= Mt) return;                  // wave-uniform exit
    const int half = lane >> 4;
    const int l15  = lane & 15;
    const int rA   = stripe * 16 + l15;

    v8f acc[8];
    {
        v8f zero = {};
#pragma unroll
        for (int t = 0; t < 8; ++t) acc[t] = zero;
    }

#pragma unroll
    for (int ks = 0; ks < KSTEPS; ++ks) {
        union { uint4 u[2]; v16bf v; } af;
        const int kb = (ks << 5) + (half << 3);
        const uint4* ap = (const uint4*)(A + (size_t)rA * K0 + kb);
        af.u[0] = ap[0];       // K = kb .. kb+7
        af.u[1] = ap[2];       // K = kb+16 .. kb+23
        const int kr = (ks << 5) + (half << 3);
        const unsigned short* wp = Wa + (size_t)kr * Hc + l15;
#pragma unroll
        for (int tn = 0; tn < 8; ++tn) {
            union { uint4 u[2]; unsigned short e[16]; v16bf v; } bv;
#pragma unroll
            for (int s = 0; s < 16; ++s)
                bv.e[s] = wp[(size_t)(s < 8 ? s : s + 8) * Hc + tn * 16];
            acc[tn] = __builtin_amdgcn_wmma_f32_16x16x32_bf16(false, af.v, false, bv.v,
                                                              (short)0, acc[tn], false, false);
        }
    }
    if (HASA2) {
#pragma unroll
        for (int ks = 0; ks < 4; ++ks) {       // second pair always K=128
            union { uint4 u[2]; v16bf v; } af;
            const int kb = (ks << 5) + (half << 3);
            const uint4* ap = (const uint4*)(A2 + (size_t)rA * Hc + kb);
            af.u[0] = ap[0];
            af.u[1] = ap[2];
            const int kr = (ks << 5) + (half << 3);
            const unsigned short* wp = Wb + (size_t)kr * Hc + l15;
#pragma unroll
            for (int tn = 0; tn < 8; ++tn) {
                union { uint4 u[2]; unsigned short e[16]; v16bf v; } bv;
#pragma unroll
                for (int s = 0; s < 16; ++s)
                    bv.e[s] = wp[(size_t)(s < 8 ? s : s + 8) * Hc + tn * 16];
                acc[tn] = __builtin_amdgcn_wmma_f32_16x16x32_bf16(false, af.v, false, bv.v,
                                                                  (short)0, acc[tn], false, false);
            }
        }
    }

    // Row-gather indices for the fused add (one load per output row)
    int ar[8];
    if (addRows) {
#pragma unroll
        for (int r = 0; r < 8; ++r) {
            const int row = stripe * 16 + r + (half << 3);
            ar[r] = (row < M) ? (addIdx ? addIdx[row] : row) : 0;
        }
    }

#pragma unroll
    for (int tn = 0; tn < 8; ++tn) {
        const int col = (tn << 4) + l15;
        const float bvv = bias ? bias[col] : 0.f;
#pragma unroll
        for (int r = 0; r < 8; ++r) {
            const int row = stripe * 16 + r + (half << 3);
            if (row < M) {
                float v = acc[tn][r] + bvv;
                if (addRows) v += bfu2f(addRows[(size_t)ar[r] * Hc + col]);
                if (doRelu) v = fmaxf(v, 0.f);
                out[(size_t)row * Hc + col] = f2bfu(v);
            }
        }
    }
}

__global__ __launch_bounds__(256) void fill0_k(float* __restrict__ p, long long n) {
    long long t = (long long)blockIdx.x * blockDim.x + threadIdx.x;
    if (t < n) p[t] = 0.f;
}

__global__ __launch_bounds__(256) void fillh_k(unsigned short* __restrict__ p, long long n) {
    long long t = (long long)blockIdx.x * blockDim.x + threadIdx.x;
    if (t < n) p[t] = 0;
}

// f32 -> bf16 conversion (weights, x0)
__global__ __launch_bounds__(256) void cvt_k(const float* __restrict__ in,
                                             unsigned short* __restrict__ out, long long n) {
    long long t = (long long)blockIdx.x * blockDim.x + threadIdx.x;
    if (t < n) out[t] = f2bfu(in[t]);
}

// sum[ids[r]] += bf2f(x[gidx?gidx[r]:r]); cnt[ids[r]] += 1.  One thread = (row, 4 feats).
__global__ __launch_bounds__(256) void scatter_add_k(
    const unsigned short* __restrict__ x, const int* __restrict__ gidx,
    const int* __restrict__ ids, int rows,
    float* __restrict__ sum, float* __restrict__ cnt)
{
    long long t = (long long)blockIdx.x * blockDim.x + threadIdx.x;
    int row = (int)(t >> 5);
    if (row >= rows) return;
    int c = ((int)t & 31) * 4;
    int s = gidx ? gidx[row] : row;
    int d = ids[row];
    const unsigned short* xp = x + (size_t)s * Hc + c;
    float* o = sum + (size_t)d * Hc + c;
    atomicAdd(o + 0, bfu2f(xp[0]));
    atomicAdd(o + 1, bfu2f(xp[1]));
    atomicAdd(o + 2, bfu2f(xp[2]));
    atomicAdd(o + 3, bfu2f(xp[3]));
    if (c == 0) atomicAdd(cnt + d, 1.0f);
}

// dst[i*stride+f] = bf16( sum[i*128+f] / max(cnt[i],1) )
__global__ __launch_bounds__(256) void seg_div_k(
    const float* __restrict__ sum, const float* __restrict__ cnt,
    unsigned short* __restrict__ dst, int nseg, int dstStride)
{
    int t = blockIdx.x * blockDim.x + threadIdx.x;
    if (t >= nseg * Hc) return;
    int i = t >> 7, f = t & 127;
    float c = cnt ? fmaxf(cnt[i], 1.0f) : 1.0f;
    dst[(size_t)i * dstStride + f] = f2bfu(sum[(size_t)i * Hc + f] / c);
}

// bf16 copy with row re-stride (h2 -> g slice)
__global__ __launch_bounds__(256) void copy_bf_k(
    const unsigned short* __restrict__ src, unsigned short* __restrict__ dst,
    int nseg, int dstStride)
{
    int t = blockIdx.x * blockDim.x + threadIdx.x;
    if (t >= nseg * Hc) return;
    int i = t >> 7, f = t & 127;
    dst[(size_t)i * dstStride + f] = src[(size_t)i * Hc + f];
}

__global__ __launch_bounds__(256) void make_assign0_k(
    const int* __restrict__ b0, const int* __restrict__ a0l, int* __restrict__ out, int n)
{
    int t = blockIdx.x * blockDim.x + threadIdx.x;
    if (t < n) out[t] = a0l[t] + b0[t] * Ff;
}

__global__ __launch_bounds__(256) void make_assign1_k(int* __restrict__ out) {
    int t = blockIdx.x * blockDim.x + threadIdx.x;
    if (t < TFf) out[t] = t / Ff;
}

// out[b,c] = hb2[c] + z[b,:] . H2[:,c]   (z bf16, H2 [128][2] f32 row-major)
__global__ __launch_bounds__(128) void head2_k(
    const unsigned short* __restrict__ z, const float* __restrict__ W,
    const float* __restrict__ bias, float* __restrict__ out)
{
    int t = blockIdx.x * blockDim.x + threadIdx.x;
    if (t >= Bb * Cc) return;
    int b = t / Cc, c = t % Cc;
    float s = bias[c];
#pragma unroll 8
    for (int k = 0; k < Hc; ++k) s += bfu2f(z[b * Hc + k]) * W[k * Cc + c];
    out[t] = s;
}

extern "C" void kernel_launch(void* const* d_in, const int* in_sizes, int n_in,
                              void* d_out, int out_size, void* d_ws, size_t ws_size,
                              hipStream_t stream)
{
    (void)in_sizes; (void)n_in; (void)out_size; (void)ws_size;
    const float* x0   = (const float*)d_in[0];
    const int*   eidx = (const int*)d_in[1];
    const int*   src  = eidx;
    const int*   dstE = eidx + Ee;
    const int*   b0   = (const int*)d_in[2];
    const int*   a0l  = (const int*)d_in[3];
    const float* fWp   = (const float*)d_in[4];
    const float* bp    = (const float*)d_in[5];
    const float* fe0Ws = (const float*)d_in[6];
    const float* fe0Wn = (const float*)d_in[7];
    const float* e0b   = (const float*)d_in[8];
    const float* fc0Ws = (const float*)d_in[9];
    const float* fc0Wn = (const float*)d_in[10];
    const float* c0b   = (const float*)d_in[11];
    const float* fe1Ws = (const float*)d_in[12];
    const float* e1b   = (const float*)d_in[13];
    const float* fe2Ws = (const float*)d_in[14];
    const float* e2b   = (const float*)d_in[15];
    const float* fc1Ws = (const float*)d_in[16];
    const float* c1b   = (const float*)d_in[17];
    const float* fc2Ws = (const float*)d_in[18];
    const float* c2b   = (const float*)d_in[19];
    const float* fA0 = (const float*)d_in[20];
    const float* fA1 = (const float*)d_in[21];
    const float* fA2 = (const float*)d_in[22];
    const float* fU0 = (const float*)d_in[23];
    const float* fU1 = (const float*)d_in[24];
    const float* fD1 = (const float*)d_in[25];
    const float* fD2 = (const float*)d_in[26];
    const float* ib0 = (const float*)d_in[27];
    const float* ib1 = (const float*)d_in[28];
    const float* ib2 = (const float*)d_in[29];
    const float* fH1 = (const float*)d_in[30];
    const float* hb1 = (const float*)d_in[31];
    const float* H2w = (const float*)d_in[32];
    const float* hb2 = (const float*)d_in[33];
    float* outp = (float*)d_out;

    // ---- bump allocator over d_ws ----
    char* base = (char*)d_ws; size_t off = 0;
    auto alloc = [&](size_t bytes) -> void* {
        off = (off + 255) & ~(size_t)255;
        void* p = base + off; off += bytes; return p;
    };
    typedef unsigned short bfu;
    const size_t HH = (size_t)Hc * Hc;

    int* assign0 = (int*)alloc((size_t)Nn * 4);
    int* assign1 = (int*)alloc((size_t)TFf * 4);
    // bf16 activation/state buffers (B-scale buffers padded to Bpad rows)
    bfu* x0b  = (bfu*)alloc((size_t)Nn * D0c * 2);
    bfu* h0a  = (bfu*)alloc((size_t)Nn * Hc * 2);
    bfu* h0b  = (bfu*)alloc((size_t)Nn * Hc * 2);
    bfu* agg0 = (bfu*)alloc((size_t)Nn * Hc * 2);
    bfu* h1a  = (bfu*)alloc((size_t)TFf * Hc * 2);
    bfu* h1b  = (bfu*)alloc((size_t)TFf * Hc * 2);
    bfu* h2a  = (bfu*)alloc((size_t)Bpad * Hc * 2);
    bfu* h2b  = (bfu*)alloc((size_t)Bpad * Hc * 2);
    bfu* p01  = (bfu*)alloc((size_t)TFf * Hc * 2);
    bfu* p12  = (bfu*)alloc((size_t)Bpad * Hc * 2);
    bfu* t1   = (bfu*)alloc((size_t)TFf * Hc * 2);
    bfu* t2   = (bfu*)alloc((size_t)Bpad * Hc * 2);
    bfu* g    = (bfu*)alloc((size_t)Bpad * 3 * Hc * 2);
    bfu* z    = (bfu*)alloc((size_t)Bpad * Hc * 2);
    // f32 accumulation buffers for segment sums
    float* sum0 = (float*)alloc((size_t)Nn * Hc * 4);
    float* cnt0 = (float*)alloc((size_t)Nn * 4);
    float* sum1 = (float*)alloc((size_t)TFf * Hc * 4);
    float* cnt1 = (float*)alloc((size_t)TFf * 4);
    float* sumB = (float*)alloc((size_t)Bb * Hc * 4);
    float* cntB = (float*)alloc((size_t)Bb * 4);
    // bf16 weights
    bfu* Wp  = (bfu*)alloc((size_t)D0c * Hc * 2);
    bfu* e0Ws = (bfu*)alloc(2 * HH * 2); bfu* e0Wn = (bfu*)alloc(2 * HH * 2);
    bfu* c0Ws = (bfu*)alloc(2 * HH * 2); bfu* c0Wn = (bfu*)alloc(2 * HH * 2);
    bfu* e1Ws = (bfu*)alloc(2 * HH * 2); bfu* e2Ws = (bfu*)alloc(2 * HH * 2);
    bfu* c1Ws = (bfu*)alloc(2 * HH * 2); bfu* c2Ws = (bfu*)alloc(2 * HH * 2);
    bfu* A0w = (bfu*)alloc(HH * 2); bfu* A1w = (bfu*)alloc(HH * 2); bfu* A2w = (bfu*)alloc(HH * 2);
    bfu* U0w = (bfu*)alloc(HH * 2); bfu* U1w = (bfu*)alloc(HH * 2);
    bfu* D1w = (bfu*)alloc(HH * 2); bfu* D2w = (bfu*)alloc(HH * 2);
    bfu* H1w = (bfu*)alloc((size_t)3 * Hc * Hc * 2);

    auto cvt = [&](const float* in, bfu* out, long long n) {
        cvt_k<<<(unsigned)((n + 255) / 256), 256, 0, stream>>>(in, out, n);
    };
    auto fill0 = [&](float* p, long long n) {
        fill0_k<<<(unsigned)((n + 255) / 256), 256, 0, stream>>>(p, n);
    };
    auto fillh = [&](bfu* p, long long n) {
        fillh_k<<<(unsigned)((n + 255) / 256), 256, 0, stream>>>(p, n);
    };
    auto gemm = [&](const bfu* A, const bfu* Wa, int K0,
                    const bfu* A2p, const bfu* Wbp,
                    const bfu* addR, const int* addI,
                    const float* bias, bfu* o, int M, int relu) {
        unsigned Mt = (unsigned)((M + 15) / 16);
        unsigned blocks = (Mt + 7) / 8;
        if (K0 == 64)
            gemm_stripe_k<2, false><<<blocks, 256, 0, stream>>>(A, Wa, nullptr, nullptr, addR, addI, bias, o, M, relu);
        else if (K0 == 384)
            gemm_stripe_k<12, false><<<blocks, 256, 0, stream>>>(A, Wa, nullptr, nullptr, addR, addI, bias, o, M, relu);
        else if (A2p)
            gemm_stripe_k<4, true><<<blocks, 256, 0, stream>>>(A, Wa, A2p, Wbp, addR, addI, bias, o, M, relu);
        else
            gemm_stripe_k<4, false><<<blocks, 256, 0, stream>>>(A, Wa, nullptr, nullptr, addR, addI, bias, o, M, relu);
    };
    auto seg_mean = [&](const bfu* x, const int* gidx, const int* ids, int rows,
                        float* sum, float* cnt, int nseg, bfu* dstp, int dstStride) {
        fill0(sum, (long long)nseg * Hc);
        fill0(cnt, nseg);
        long long thr = (long long)rows * 32;
        scatter_add_k<<<(unsigned)((thr + 255) / 256), 256, 0, stream>>>(x, gidx, ids, rows, sum, cnt);
        int tt = nseg * Hc;
        seg_div_k<<<(unsigned)((tt + 255) / 256), 256, 0, stream>>>(sum, cnt, dstp, nseg, dstStride);
    };

    // ---- one-time conversions ----
    cvt(x0, x0b, (long long)Nn * D0c);
    cvt(fWp, Wp, (long long)D0c * Hc);
    cvt(fe0Ws, e0Ws, 2 * (long long)HH); cvt(fe0Wn, e0Wn, 2 * (long long)HH);
    cvt(fc0Ws, c0Ws, 2 * (long long)HH); cvt(fc0Wn, c0Wn, 2 * (long long)HH);
    cvt(fe1Ws, e1Ws, 2 * (long long)HH); cvt(fe2Ws, e2Ws, 2 * (long long)HH);
    cvt(fc1Ws, c1Ws, 2 * (long long)HH); cvt(fc2Ws, c2Ws, 2 * (long long)HH);
    cvt(fA0, A0w, (long long)HH); cvt(fA1, A1w, (long long)HH); cvt(fA2, A2w, (long long)HH);
    cvt(fU0, U0w, (long long)HH); cvt(fU1, U1w, (long long)HH);
    cvt(fD1, D1w, (long long)HH); cvt(fD2, D2w, (long long)HH);
    cvt(fH1, H1w, (long long)3 * Hc * Hc);

    make_assign0_k<<<(Nn + 255) / 256, 256, 0, stream>>>(b0, a0l, assign0, Nn);
    make_assign1_k<<<(TFf + 255) / 256, 256, 0, stream>>>(assign1);

    bfu *cur0 = h0a, *alt0 = h0b;
    bfu *cur1 = h1a, *alt1 = h1b;
    bfu *cur2 = h2a, *alt2 = h2b;

    // h0 = relu(x0 @ Wp + bp)
    gemm(x0b, Wp, D0c, nullptr, nullptr, nullptr, nullptr, bp, cur0, Nn, 1);
    fillh(cur1, (long long)TFf * Hc);
    fillh(cur2, (long long)Bpad * Hc);

    // e0in: 2-layer edge encoder on scale 0
    for (int l = 0; l < 2; ++l) {
        seg_mean(cur0, src, dstE, Ee, sum0, cnt0, Nn, agg0, Hc);
        gemm(cur0, e0Ws + (size_t)l * HH, Hc, agg0, e0Wn + (size_t)l * HH,
             nullptr, nullptr, e0b + (size_t)l * Hc, alt0, Nn, 1);
        std::swap(cur0, alt0);
    }
    for (int l = 0; l < 2; ++l) {
        gemm(cur1, e1Ws + (size_t)l * HH, Hc, nullptr, nullptr,
             nullptr, nullptr, e1b + (size_t)l * Hc, alt1, TFf, 1);
        std::swap(cur1, alt1);
    }
    for (int l = 0; l < 2; ++l) {
        gemm(cur2, e2Ws + (size_t)l * HH, Hc, nullptr, nullptr,
             nullptr, nullptr, e2b + (size_t)l * Hc, alt2, Bb, 1);
        std::swap(cur2, alt2);
    }

    // ---- inter-scale fusion steps ----
    for (int step = 0; step < 2; ++step) {
        seg_mean(cur0, nullptr, assign0, Nn, sum1, cnt1, TFf, p01, Hc);   // p01
        seg_mean(cur1, nullptr, assign1, TFf, sumB, cntB, Bb, p12, Hc);   // p12
        gemm(cur1, D1w, Hc, nullptr, nullptr, nullptr, nullptr, nullptr, t1, TFf, 0);
        gemm(cur2, D2w, Hc, nullptr, nullptr, nullptr, nullptr, nullptr, t2, Bb, 0);
        // n0 = relu(h0@A0 + t1[assign0] + ib0)
        gemm(cur0, A0w, Hc, nullptr, nullptr, t1, assign0, ib0, alt0, Nn, 1);
        // n1 = relu(h1@A1 + p01@U0 + t2[assign1] + ib1)
        gemm(cur1, A1w, Hc, p01, U0w, t2, assign1, ib1, alt1, TFf, 1);
        // n2 = relu(h2@A2 + p12@U1 + ib2)
        gemm(cur2, A2w, Hc, p12, U1w, nullptr, nullptr, ib2, alt2, Bb, 1);
        std::swap(cur0, alt0); std::swap(cur1, alt1); std::swap(cur2, alt2);
        for (int l = 0; l < 2; ++l) {
            seg_mean(cur0, src, dstE, Ee, sum0, cnt0, Nn, agg0, Hc);
            gemm(cur0, c0Ws + (size_t)l * HH, Hc, agg0, c0Wn + (size_t)l * HH,
                 nullptr, nullptr, c0b + (size_t)l * Hc, alt0, Nn, 1);
            std::swap(cur0, alt0);
        }
        for (int l = 0; l < 2; ++l) {
            gemm(cur1, c1Ws + (size_t)l * HH, Hc, nullptr, nullptr,
                 nullptr, nullptr, c1b + (size_t)l * Hc, alt1, TFf, 1);
            std::swap(cur1, alt1);
        }
        for (int l = 0; l < 2; ++l) {
            gemm(cur2, c2Ws + (size_t)l * HH, Hc, nullptr, nullptr,
                 nullptr, nullptr, c2b + (size_t)l * Hc, alt2, Bb, 1);
            std::swap(cur2, alt2);
        }
    }

    // ---- readout: g = [g0 | g1 | h2] in [B, 3H] ----
    seg_mean(cur0, nullptr, b0,      Nn,  sumB, cntB, Bb, g,      3 * Hc);  // g0
    seg_mean(cur1, nullptr, assign1, TFf, sumB, cntB, Bb, g + Hc, 3 * Hc);  // g1
    {
        int tt = Bb * Hc;
        copy_bf_k<<<(unsigned)((tt + 255) / 256), 256, 0, stream>>>(cur2, g + 2 * Hc, Bb, 3 * Hc);
    }
    // z = relu(g @ H1 + hb1)
    gemm(g, H1w, 3 * Hc, nullptr, nullptr, nullptr, nullptr, hb1, z, Bb, 1);
    // out = z @ H2 + hb2
    head2_k<<<1, 128, 0, stream>>>(z, H2w, hb2, outp);
}
